// LinearAttention_50964081934578
// MI455X (gfx1250) — compile-verified, matrix operand order checked by reading, not run on hardware
//
#include <hip/hip_runtime.h>
#include <hip/hip_bf16.h>

// ---------------------------------------------------------------------------
// LinearAttention for MI455X (gfx1250, wave32, WMMA).
// bf16 matmuls with f32 accumulation via v_wmma_f32_16x16x32_bf16.
// GEMM uses 128x128 macro-tiles, double-buffered LDS fed by
// GLOBAL_LOAD_ASYNC_TO_LDS_B128 (ASYNCcnt) when the builtin is available.
// Weights pre-transposed (N-major) so A and B fragments are contiguous
// 16B ds_load_b128 reads. Assumes ws_size >= ~140 MB.
// ---------------------------------------------------------------------------

typedef __attribute__((ext_vector_type(16))) __bf16 v16bf;
typedef __attribute__((ext_vector_type(8)))  float  v8f;
typedef __attribute__((ext_vector_type(4)))  int    v4i;

#define HID   2048
#define NH    16
#define HD    128
#define BATCH 2
#define SEQ   2048
#define MTOK  (BATCH * SEQ)   // 4096 token rows

#if defined(__gfx1250__) && \
    __has_builtin(__builtin_amdgcn_global_load_async_to_lds_b128) && \
    __has_builtin(__builtin_amdgcn_s_wait_asynccnt)
#define HAS_ASYNC 1
typedef __attribute__((address_space(1))) v4i as1_v4i;  // global v4i
typedef __attribute__((address_space(3))) v4i as3_v4i;  // LDS v4i
#else
#define HAS_ASYNC 0
#endif

// 16-byte global -> LDS copy. Async (no VGPR transit, ASYNCcnt) when possible.
static __device__ inline void copy16_g2l(void* lds, const void* g) {
#if HAS_ASYNC
  void* gm = const_cast<void*>(g);
  __builtin_amdgcn_global_load_async_to_lds_b128(
      (as1_v4i*)(v4i*)gm, (as3_v4i*)(v4i*)lds, 0, 0);
#else
  *(uint4*)lds = *(const uint4*)g;
#endif
}
static __device__ inline void wait_g2l() {
#if HAS_ASYNC
  __builtin_amdgcn_s_wait_asynccnt(0);
#endif
}

static __device__ inline v8f wmma_bf16(v16bf a, v16bf b, v8f c) {
  // (neg_a, A, neg_b, B, c_mod, C, reuse_a, reuse_b)
  return __builtin_amdgcn_wmma_f32_16x16x32_bf16(false, a, false, b,
                                                 (short)0, c, false, false);
}

union FragU { v16bf v; uint4 q[2]; };

// Load a 16x32 bf16 fragment from a row-major [rows][32] LDS tile.
// ISA 16-bit A layout (wave32): lanes 0-15 -> m=lane, k-base 0;
// lanes 16-31 -> m=lane-16, k-base 8; dwords 0-3: k = kb..kb+7 (contiguous),
// dwords 4-7: k = 16+kb..16+kb+7 (contiguous). Two ds_load_b128 per fragment.
// B fragments (n-major staging) use the mirrored layout with lane -> n.
static __device__ inline v16bf load_frag(const __bf16* lds, int rowStart, int lane) {
  const int half = lane >> 4;
  const int m    = lane & 15;
  const __bf16* row = lds + (rowStart + m) * 32;
  FragU f;
  f.q[0] = *(const uint4*)(row + half * 8);
  f.q[1] = *(const uint4*)(row + 16 + half * 8);
  return f.v;
}

// ---------------------------------------------------------------------------
// Generic bf16 WMMA GEMM:  C[M,N] = A[M,K] * Bt[N,K]^T  (+bias) (*scale) (relu)
// Macro tile 128(M) x 128(N), K-step 32, double-buffered async LDS staging.
// 256 threads = 8 waves (4x2); each wave owns 32x64 = 2x4 WMMA accumulators.
// blockIdx.z selects a "slice" via element strides (heads / batched GEMMs).
// ---------------------------------------------------------------------------
__global__ __launch_bounds__(256)
void gemm_bf16_wmma(const __bf16* __restrict__ A, int lda, long strideAz,
                    const __bf16* __restrict__ Bt, int ldb, long strideBz,
                    float* __restrict__ Cf, __bf16* __restrict__ Cb,
                    int ldc, long strideCz,
                    const float* __restrict__ bias,
                    const float* __restrict__ denom, int denomBase,
                    int M, int N, int K, int relu) {
  __shared__ __attribute__((aligned(16))) __bf16 As[2][128][32];  // 2 x 8 KB
  __shared__ __attribute__((aligned(16))) __bf16 Bs[2][128][32];  // 2 x 8 KB

  const int z = blockIdx.z;
  A  += (long)z * strideAz;
  Bt += (long)z * strideBz;
  const long cOff = (long)z * strideCz;

  const int tid  = threadIdx.x;
  const int lane = tid & 31;
  const int wave = tid >> 5;
  const int wm   = (wave >> 1) * 32;   // 0..96
  const int wn   = (wave & 1) * 64;    // 0 / 64

  const int blockM = blockIdx.y * 128;
  const int blockN = blockIdx.x * 128;

  v8f acc[2][4] = {};

  // Cooperative staging: 512 16B chunks per tile; each thread moves chunks
  // {tid, tid+256} of A and of B.  chunk -> (row = cid>>2, col = (cid&3)*8).
  const int r0 = tid >> 2;             // 0..63
  const int r1 = r0 + 64;              // 64..127
  const int c0 = (tid & 3) * 8;        // 0..24

  auto stage = [&](int k0, int buf) {
    copy16_g2l(&As[buf][r0][c0], A  + (long)(blockM + r0) * lda + k0 + c0);
    copy16_g2l(&As[buf][r1][c0], A  + (long)(blockM + r1) * lda + k0 + c0);
    copy16_g2l(&Bs[buf][r0][c0], Bt + (long)(blockN + r0) * ldb + k0 + c0);
    copy16_g2l(&Bs[buf][r1][c0], Bt + (long)(blockN + r1) * ldb + k0 + c0);
  };

  stage(0, 0);
  int buf = 0;
  for (int k0 = 0; k0 < K; k0 += 32, buf ^= 1) {
    wait_g2l();          // this thread's chunks of `buf` have landed
    __syncthreads();     // everyone's chunks visible; buf^1 no longer being read
    if (k0 + 32 < K) stage(k0 + 32, buf ^ 1);  // overlap with WMMA below

    const v16bf a0 = load_frag(&As[buf][0][0], wm,      lane);
    const v16bf a1 = load_frag(&As[buf][0][0], wm + 16, lane);
    v16bf bf4[4];
#pragma unroll
    for (int t = 0; t < 4; ++t) bf4[t] = load_frag(&Bs[buf][0][0], wn + t * 16, lane);
#pragma unroll
    for (int tn = 0; tn < 4; ++tn) {
      acc[0][tn] = wmma_bf16(a0, bf4[tn], acc[0][tn]);
      acc[1][tn] = wmma_bf16(a1, bf4[tn], acc[1][tn]);
    }
  }

  float scale = 1.0f;
  if (denom) scale = 1.0f / (denom[denomBase + z] + 1e-8f);
  const int half = lane >> 4;

  for (int tm = 0; tm < 2; ++tm)
    for (int tn = 0; tn < 4; ++tn) {
      const int col   = blockN + wn + tn * 16 + (lane & 15);
      const int rbase = blockM + wm + tm * 16 + half * 8;
      const float bv  = bias ? bias[col] : 0.0f;
#pragma unroll
      for (int j = 0; j < 8; ++j) {   // C layout: VGPR j -> M = (half?8:0)+j
        float v = acc[tm][tn][j] * scale + bv;
        if (relu) v = fmaxf(v, 0.0f);
        const long idx = cOff + (long)(rbase + j) * ldc + col;
        if (Cf) Cf[idx] = v;
        else    Cb[idx] = (__bf16)v;
      }
    }
}

// ---------------------------------------------------------------------------
// KVt[z][e][d] = sum_s V[b,s,h*128+e] * Kf[b,s,h*128+d]   (= (K^T V)^T)
// One block per (b,h). LDS-transposed staging of both 32x128 s-slabs so WMMA
// fragments are contiguous. 8 waves (4x2): 32(e) x 64(d) per wave.
// ---------------------------------------------------------------------------
__global__ __launch_bounds__(256)
void kvt_bf16_wmma(const __bf16* __restrict__ Vb,   // [4096][2048]
                   const __bf16* __restrict__ Kf,   // [4096][2048]
                   __bf16* __restrict__ KVt) {      // [32][128][128]
  __shared__ __attribute__((aligned(16))) __bf16 At[128][32];  // V^T slab
  __shared__ __attribute__((aligned(16))) __bf16 Bt[128][32];  // K^T slab

  const int z = blockIdx.x;            // b*16 + h
  const int b = z >> 4, h = z & 15;
  const __bf16* Vbase = Vb + (long)b * SEQ * HID + h * HD;
  const __bf16* Kbase = Kf + (long)b * SEQ * HID + h * HD;

  const int tid = threadIdx.x, lane = tid & 31, wave = tid >> 5;
  const int wm = (wave >> 1) * 32;     // e rows: 0..96
  const int wn = (wave & 1) * 64;      // d cols: 0 / 64
  v8f acc[2][4] = {};

  const int srow = tid >> 3;           // 0..31
  const int coff = (tid & 7) * 16;     // 0..112

  for (int s0 = 0; s0 < SEQ; s0 += 32) {
    union { uint4 q[2]; __bf16 e[16]; } va, ka;
    const __bf16* vp = Vbase + (long)(s0 + srow) * HID + coff;
    const __bf16* kp = Kbase + (long)(s0 + srow) * HID + coff;
    va.q[0] = *(const uint4*)vp;       va.q[1] = *(const uint4*)(vp + 8);
    ka.q[0] = *(const uint4*)kp;       ka.q[1] = *(const uint4*)(kp + 8);
    if (s0 + 32 < SEQ) {               // warm L2/L0 for the next slab
      __builtin_prefetch(vp + 32 * HID, 0, 3);
      __builtin_prefetch(kp + 32 * HID, 0, 3);
    }
    __syncthreads();
#pragma unroll
    for (int j = 0; j < 16; ++j) {     // transpose into LDS
      At[coff + j][srow] = va.e[j];
      Bt[coff + j][srow] = ka.e[j];
    }
    __syncthreads();

    v16bf af[2], bf4[4];
    af[0] = load_frag(&At[0][0], wm,      lane);
    af[1] = load_frag(&At[0][0], wm + 16, lane);
#pragma unroll
    for (int t = 0; t < 4; ++t) bf4[t] = load_frag(&Bt[0][0], wn + t * 16, lane);
#pragma unroll
    for (int tm = 0; tm < 2; ++tm)
#pragma unroll
      for (int tn = 0; tn < 4; ++tn)
        acc[tm][tn] = wmma_bf16(af[tm], bf4[tn], acc[tm][tn]);
  }

  __bf16* out = KVt + (long)z * HD * HD;
  const int half = lane >> 4;
  for (int tm = 0; tm < 2; ++tm)
    for (int tn = 0; tn < 4; ++tn) {
      const int col   = wn + tn * 16 + (lane & 15);
      const int rbase = wm + tm * 16 + half * 8;
#pragma unroll
      for (int j = 0; j < 8; ++j)
        out[(long)(rbase + j) * HD + col] = (__bf16)acc[tm][tn][j];
    }
}

// denom[z] = sum over Kf[b, :, h*128 : h*128+128]   (one block per (b,h))
__global__ __launch_bounds__(256)
void denom_kernel(const __bf16* __restrict__ Kf, float* __restrict__ denom) {
  __shared__ float red[256];
  const int z = blockIdx.x, b = z >> 4, h = z & 15;
  const __bf16* base = Kf + (long)b * SEQ * HID + h * HD;
  float s = 0.0f;
  for (int i = threadIdx.x; i < SEQ * HD; i += 256) {
    const int sr = i >> 7, d = i & 127;
    s += (float)base[(long)sr * HID + d];
  }
  red[threadIdx.x] = s;
  __syncthreads();
  for (int off = 128; off; off >>= 1) {
    if (threadIdx.x < off) red[threadIdx.x] += red[threadIdx.x + off];
    __syncthreads();
  }
  if (threadIdx.x == 0) denom[z] = red[0];
}

// f32 -> bf16 flat cast (n multiple of 4)
__global__ void cast_f32_bf16(const float* __restrict__ in,
                              __bf16* __restrict__ out, long n) {
  const long i = ((long)blockIdx.x * blockDim.x + threadIdx.x) * 4;
  if (i + 3 < n) {
    const float4 v = *(const float4*)(in + i);
    union { __bf16 h[4]; uint2 u; } o;
    o.h[0] = (__bf16)v.x; o.h[1] = (__bf16)v.y;
    o.h[2] = (__bf16)v.z; o.h[3] = (__bf16)v.w;
    *(uint2*)(out + i) = o.u;
  }
}

// Wt[n][k] = (bf16) W[k][n] — tiled 32x32 through LDS, coalesced both ways.
__global__ __launch_bounds__(256)
void transpose_cast(const float* __restrict__ W, __bf16* __restrict__ Wt,
                    int K, int N) {
  __shared__ float t[32][33];
  const int k0 = blockIdx.x * 32, n0 = blockIdx.y * 32;
  const int c = threadIdx.x & 31, r = threadIdx.x >> 5;
#pragma unroll
  for (int i = 0; i < 32; i += 8)
    t[r + i][c] = W[(long)(k0 + r + i) * N + n0 + c];
  __syncthreads();
#pragma unroll
  for (int i = 0; i < 32; i += 8)
    Wt[(long)(n0 + r + i) * K + k0 + c] = (__bf16)t[c][r + i];
}

// attn_weights = 1/S everywhere (pure-bandwidth 128-bit stores)
__global__ void fill_attn(float* __restrict__ out, long n, float val) {
  const long i = ((long)blockIdx.x * blockDim.x + threadIdx.x) * 4;
  if (i + 3 < n) *(float4*)(out + i) = make_float4(val, val, val, val);
}

// ---------------------------------------------------------------------------
static inline size_t alignup(size_t x) { return (x + 255) & ~(size_t)255; }

extern "C" void kernel_launch(void* const* d_in, const int* in_sizes, int n_in,
                              void* d_out, int out_size, void* d_ws, size_t ws_size,
                              hipStream_t stream) {
  (void)in_sizes; (void)n_in; (void)out_size; (void)ws_size;
  const float* q   = (const float*)d_in[0];
  const float* k   = (const float*)d_in[1];
  const float* v   = (const float*)d_in[2];
  const float* Wq  = (const float*)d_in[3];  const float* bq  = (const float*)d_in[4];
  const float* Wk  = (const float*)d_in[5];  const float* bk  = (const float*)d_in[6];
  const float* Wv  = (const float*)d_in[7];  const float* bv  = (const float*)d_in[8];
  const float* Wo  = (const float*)d_in[9];  const float* bo  = (const float*)d_in[10];
  const float* Wf1 = (const float*)d_in[11]; const float* bf1 = (const float*)d_in[12];
  const float* Wf2 = (const float*)d_in[13]; const float* bf2 = (const float*)d_in[14];
  float* out = (float*)d_out;

  // ---- workspace carving (aggressively reused; ~130 MB total) ----
  char* p = (char*)d_ws;
  auto carve = [&](size_t bytes) { void* r = (void*)p; p += alignup(bytes); return r; };
  const size_t TOKHID = (size_t)MTOK * HID;           // 8,388,608 elements
  __bf16* xb    = (__bf16*)carve(TOKHID * 2);         // cast buffer (q/k/v)
  __bf16* P0    = (__bf16*)carve(TOKHID * 2);         // proj out, later context
  __bf16* vpb   = (__bf16*)carve(TOKHID * 2);         // V projection (persists)
  __bf16* Hbuf  = (__bf16*)carve((size_t)MTOK * 2 * HID * 2);  // feature hidden
  __bf16* Qf    = (__bf16*)carve(TOKHID * 2);
  __bf16* Kf    = (__bf16*)carve(TOKHID * 2);
  __bf16* WT    = (__bf16*)carve((size_t)HID * HID * 2);       // reused 4x
  __bf16* Wf1T  = (__bf16*)carve((size_t)256 * 128 * 2);       // [256][128]
  __bf16* Wf2T  = (__bf16*)carve((size_t)128 * 256 * 2);       // [128][256]
  __bf16* KVt   = (__bf16*)carve((size_t)32 * HD * HD * 2);    // [32][128][128]
  float*  dnm   = (float*)carve(32 * sizeof(float));

  const dim3 blk(256);
  const dim3 gCast((unsigned)(TOKHID / 4 / 256));
  const dim3 gTrW(HID / 32, HID / 32);
  const dim3 gProj(HID / 128, MTOK / 128, 1);

  // Feature-map weights (shared across heads)
  transpose_cast<<<dim3(128 / 32, 256 / 32), blk, 0, stream>>>(Wf1, Wf1T, 128, 256);
  transpose_cast<<<dim3(256 / 32, 128 / 32), blk, 0, stream>>>(Wf2, Wf2T, 256, 128);

  auto proj = [&](const float* x, const float* W, const float* bias, __bf16* dst) {
    cast_f32_bf16<<<gCast, blk, 0, stream>>>(x, xb, (long)TOKHID);
    transpose_cast<<<gTrW, blk, 0, stream>>>(W, WT, HID, HID);
    gemm_bf16_wmma<<<gProj, blk, 0, stream>>>(
        xb, HID, 0, WT, HID, 0, nullptr, dst, HID, 0,
        bias, nullptr, 0, MTOK, HID, HID, 0);
  };
  auto feature = [&](const __bf16* src, __bf16* dst) {
    // H = relu(src_head @ Wf1 + bf1)  : z = head, block-diagonal via strides
    gemm_bf16_wmma<<<dim3(256 / 128, MTOK / 128, NH), blk, 0, stream>>>(
        src, HID, 128, Wf1T, 128, 0, nullptr, Hbuf, 2 * HID, 256,
        bf1, nullptr, 0, MTOK, 256, 128, 1);
    // dst = H_head @ Wf2 + bf2
    gemm_bf16_wmma<<<dim3(1, MTOK / 128, NH), blk, 0, stream>>>(
        Hbuf, 2 * HID, 256, Wf2T, 256, 0, nullptr, dst, HID, 128,
        bf2, nullptr, 0, MTOK, 128, 256, 0);
  };

  // Q chain, K chain, V projection
  proj(q, Wq, bq, P0);  feature(P0, Qf);
  proj(k, Wk, bk, P0);  feature(P0, Kf);
  proj(v, Wv, bv, vpb);

  // Per-head denominator and KV^T = V^T K
  denom_kernel<<<dim3(32), blk, 0, stream>>>(Kf, dnm);
  kvt_bf16_wmma<<<dim3(32), blk, 0, stream>>>(vpb, Kf, KVt);

  // context = (Q @ KV) / denom, per (b, h); z = head, one launch per batch
  for (int b = 0; b < BATCH; ++b) {
    gemm_bf16_wmma<<<dim3(1, SEQ / 128, NH), blk, 0, stream>>>(
        Qf + (long)b * SEQ * HID, HID, 128,
        KVt + (long)b * NH * HD * HD, HD, (long)HD * HD,
        nullptr, P0 + (long)b * SEQ * HID, HID, 128,
        nullptr, dnm, b * NH, SEQ, HD, HD, 0);
  }

  // output = context @ Wo + bo  (f32 straight into d_out)
  transpose_cast<<<gTrW, blk, 0, stream>>>(Wo, WT, HID, HID);
  gemm_bf16_wmma<<<gProj, blk, 0, stream>>>(
      P0, HID, 0, WT, HID, 0, out, nullptr, HID, 0,
      bo, nullptr, 0, MTOK, HID, HID, 0);

  // attn_weights = 1/S (bandwidth-bound fill of the output tail)
  const long nAttn = (long)BATCH * NH * SEQ * SEQ;
  fill_attn<<<dim3((unsigned)(nAttn / 4 / 256)), blk, 0, stream>>>(
      out + (long)MTOK * HID, nAttn, 1.0f / (float)SEQ);
}
